// SelfAttention_1297080123768
// MI455X (gfx1250) — matrix operand order repaired and checked by reading, not compiled, not run on hardware
//
#include <hip/hip_runtime.h>

typedef __attribute__((ext_vector_type(16))) _Float16 v16h;
typedef __attribute__((ext_vector_type(8)))  float    v8f;
typedef __attribute__((ext_vector_type(4)))  int      v4i;

union FragH { v16h v; _Float16 h[16]; uint4 q[2]; };

// ---------------------------------------------------------------------------
// CDNA5 async global->LDS copy (ASYNCcnt-tracked), with sync fallback.
// Guarded by __has_builtin so host pass / older toolchains take the fallback.
// Builtin params are (int4 addrspace(1)*, int4 addrspace(3)*, imm, imm).
// ---------------------------------------------------------------------------
#if __has_builtin(__builtin_amdgcn_global_load_async_to_lds_b128) && \
    __has_builtin(__builtin_amdgcn_s_wait_asynccnt)
#define ASYNC_LDS 1
#else
#define ASYNC_LDS 0
#endif

__device__ inline void cp16_async(void* lds, const void* gsrc) {
#if ASYNC_LDS
  __builtin_amdgcn_global_load_async_to_lds_b128(
      (__attribute__((address_space(1))) v4i*)(unsigned long long)gsrc,
      (__attribute__((address_space(3))) v4i*)(unsigned int)(unsigned long long)lds,
      0, 0);
#else
  *(uint4*)lds = *(const uint4*)gsrc;
#endif
}

__device__ inline void async_wait0() {
#if ASYNC_LDS
  __builtin_amdgcn_s_wait_asynccnt(0);
#endif
}

// ---------------------------------------------------------------------------
// Fragment loaders (wave32, V_WMMA_F32_16X16X32_F16 layouts per CDNA5 ISA 7.12.2)
// A (16x32, MxK): lane<16 -> row=lane, K {kb..kb+7, kb+16..kb+23}, kb=0; lane>=16 kb=8
// B (32x16, KxN) from K-major ("transposed") LDS tile Bt[n][k]:
//   lane<16 -> col=lane, K {kb..kb+7, kb+16..kb+23}, kb=0; lane>=16 kb=8
// C/D (16x16 f32): VGPR v, lane l: row = v + 8*(l>>4), col = l&15
// ---------------------------------------------------------------------------
__device__ inline v16h lds_a_frag(const _Float16* T, int ldt, int row0, int k0) {
  const int lane = threadIdx.x & 31;
  const int r  = row0 + (lane & 15);
  const int kb = k0 + ((lane & 16) >> 1);   // 0 or 8
  FragH f;
  f.q[0] = *(const uint4*)(&T[r * ldt + kb]);
  f.q[1] = *(const uint4*)(&T[r * ldt + kb + 16]);
  return f.v;
}

__device__ inline v16h lds_b_frag(const _Float16* T, int ldt, int col0, int k0) {
  const int lane = threadIdx.x & 31;
  const int c  = col0 + (lane & 15);
  const int kb = k0 + ((lane & 16) >> 1);   // 0 or 8
  FragH f;
  f.q[0] = *(const uint4*)(&T[c * ldt + kb]);
  f.q[1] = *(const uint4*)(&T[c * ldt + kb + 16]);
  return f.v;
}

// ---------------------------------------------------------------------------
// Elementwise f32 -> f16 convert
// ---------------------------------------------------------------------------
__global__ __launch_bounds__(256) void cvt_f16(const float* __restrict__ in,
                                               _Float16* __restrict__ out, int n) {
  int i = blockIdx.x * 256 + threadIdx.x;
  if (i < n) out[i] = (_Float16)in[i];
}

// ---------------------------------------------------------------------------
// Transpose + convert: in f32 [K][N] -> out f16 [N][K]  (32x32 LDS tiles)
// ---------------------------------------------------------------------------
__global__ __launch_bounds__(256) void transpose_cvt(const float* __restrict__ in,
                                                     _Float16* __restrict__ out,
                                                     int K, int N) {
  __shared__ _Float16 t[32][33];
  const int kb = blockIdx.x * 32, nb = blockIdx.y * 32;
  const int tx = threadIdx.x & 31, ty = threadIdx.x >> 5;   // ty in 0..7
  #pragma unroll
  for (int yy = 0; yy < 32; yy += 8)
    t[yy + ty][tx] = (_Float16)in[(size_t)(kb + yy + ty) * N + nb + tx];
  __syncthreads();
  #pragma unroll
  for (int yy = 0; yy < 32; yy += 8)
    out[(size_t)(nb + yy + ty) * K + kb + tx] = t[tx][yy + ty];
}

// ---------------------------------------------------------------------------
// WMMA GEMM: C[M][N] = A[M][K] * Bt[N][K]^T  (+bias for f32 output)
// Block 256 threads = 8 waves; block tile 128x128; wave tile 64x32; K step 64
// Tiles staged with GLOBAL_LOAD_ASYNC_TO_LDS_B128; next tile prefetched.
// ---------------------------------------------------------------------------
#define G_LDT 72   // 64 + 8 halfs padding; row = 144B (16B aligned)

template <bool F32OUT>
__global__ __launch_bounds__(256)
void gemm_tn(const _Float16* __restrict__ A, const _Float16* __restrict__ Bt,
             void* __restrict__ Cout, const float* __restrict__ bias,
             int M, int N, int Kd) {
  __shared__ __align__(16) _Float16 As[128 * G_LDT];
  __shared__ __align__(16) _Float16 Bs[128 * G_LDT];

  const int tid  = threadIdx.x;
  const int lane = tid & 31;
  const int wave = tid >> 5;
  const int mbase = blockIdx.x * 128;
  const int nbase = blockIdx.y * 128;
  const int wm = (wave >> 2) * 64;    // 0 / 64
  const int wn = (wave & 3) * 32;     // 0 / 32 / 64 / 96

  v8f acc[4][2];
  #pragma unroll
  for (int i = 0; i < 4; ++i)
    #pragma unroll
    for (int j = 0; j < 2; ++j)
      #pragma unroll
      for (int e = 0; e < 8; ++e) acc[i][j][e] = 0.f;

  for (int k0 = 0; k0 < Kd; k0 += 64) {
    __syncthreads();
    #pragma unroll
    for (int it = 0; it < 4; ++it) {             // 1024 uint4 per tile / 256 thr
      const int e = tid + it * 256;
      const int r = e >> 3, q = e & 7;           // 8 uint4 per 64-half row
      cp16_async(&As[r * G_LDT + q * 8],
                 &A[(size_t)(mbase + r) * Kd + k0 + q * 8]);
      cp16_async(&Bs[r * G_LDT + q * 8],
                 &Bt[(size_t)(nbase + r) * Kd + k0 + q * 8]);
      if (k0 + 64 < Kd) {                        // prefetch next K tile
        __builtin_prefetch(&A[(size_t)(mbase + r) * Kd + k0 + 64 + q * 8], 0, 0);
        __builtin_prefetch(&Bt[(size_t)(nbase + r) * Kd + k0 + 64 + q * 8], 0, 0);
      }
    }
    async_wait0();
    __syncthreads();

    v16h af[2][4], bf[2][2];
    #pragma unroll
    for (int c = 0; c < 2; ++c) {
      #pragma unroll
      for (int i = 0; i < 4; ++i) af[c][i] = lds_a_frag(As, G_LDT, wm + i * 16, c * 32);
      #pragma unroll
      for (int j = 0; j < 2; ++j) bf[c][j] = lds_b_frag(Bs, G_LDT, wn + j * 16, c * 32);
    }
    #pragma unroll
    for (int c = 0; c < 2; ++c)
      #pragma unroll
      for (int i = 0; i < 4; ++i)
        #pragma unroll
        for (int j = 0; j < 2; ++j)
          acc[i][j] = __builtin_amdgcn_wmma_f32_16x16x32_f16(
              false, af[c][i], false, bf[c][j], (short)0, acc[i][j], false, false);
  }

  const int rlo = (lane >> 4) * 8;
  const int col = lane & 15;
  #pragma unroll
  for (int i = 0; i < 4; ++i)
    #pragma unroll
    for (int j = 0; j < 2; ++j) {
      const int n = nbase + wn + j * 16 + col;
      #pragma unroll
      for (int v = 0; v < 8; ++v) {
        const int m = mbase + wm + i * 16 + rlo + v;
        const float val = acc[i][j][v];
        if (F32OUT)
          ((float*)Cout)[(size_t)m * N + n] = val + bias[n];
        else
          ((_Float16*)Cout)[(size_t)m * N + n] = (_Float16)val;
      }
    }
}

// ---------------------------------------------------------------------------
// Flash attention: Q,K,V f16 [B,S,H*64] -> O f16 [B,S,H*64]
// 8 waves/block, wave = 16 Q rows (block = 128), key tile = 64, online softmax
// K tile staged via async-to-LDS; V transposed through VGPRs (reshape).
// ---------------------------------------------------------------------------
#define A_LDK 72   // 64 + 8 halfs; row = 144B (16B aligned)
#define SEQ   2048
#define INNER 1024

__global__ __launch_bounds__(256)
void flash_attn(const _Float16* __restrict__ Qg, const _Float16* __restrict__ Kg,
                const _Float16* __restrict__ Vg, _Float16* __restrict__ Og) {
  __shared__ __align__(16) _Float16 Ks[64 * A_LDK];        // [key][d]  (= Bt for QK^T)
  __shared__ __align__(16) _Float16 Vt[64 * A_LDK];        // [d][key]  (= Bt for PV)
  __shared__ __align__(16) _Float16 Pb[8 * 16 * A_LDK];    // per-wave P [16][64]

  const int tid  = threadIdx.x;
  const int lane = tid & 31;
  const int wave = tid >> 5;
  const int qt = blockIdx.x & 15;        // S / 128 = 16 q-tiles
  const int bh = blockIdx.x >> 4;
  const int b  = bh >> 4;
  const int hd = bh & 15;

  const size_t base = (size_t)b * SEQ * INNER + (size_t)hd * 64;
  const int qrow0 = qt * 128 + wave * 16;

  // Q rows for this wave as pre-scaled A-fragments (2 chunks over d=64)
  v16h qf[2];
  {
    const int r = qrow0 + (lane & 15);
    const _Float16* qptr = Qg + base + (size_t)r * INNER;
    #pragma unroll
    for (int c = 0; c < 2; ++c) {
      const int kb = c * 32 + ((lane & 16) >> 1);
      FragH f;
      f.q[0] = *(const uint4*)(qptr + kb);
      f.q[1] = *(const uint4*)(qptr + kb + 16);
      #pragma unroll
      for (int e = 0; e < 16; ++e) f.h[e] = f.h[e] * (_Float16)0.125f;  // * Dh^-0.5
      qf[c] = f.v;
    }
  }

  float mrow[8], lrow[8];
  v8f o[4];
  #pragma unroll
  for (int v = 0; v < 8; ++v) { mrow[v] = -1e30f; lrow[v] = 0.f; }
  #pragma unroll
  for (int j = 0; j < 4; ++j)
    #pragma unroll
    for (int v = 0; v < 8; ++v) o[j][v] = 0.f;

  _Float16* myP = Pb + wave * 16 * A_LDK;

  for (int kt = 0; kt < SEQ; kt += 64) {
    __syncthreads();
    #pragma unroll
    for (int it = 0; it < 2; ++it) {             // 64 rows * 8 uint4 / 256 thr
      const int e = tid + it * 256;
      const int r = e >> 3, q = e & 7;
      cp16_async(&Ks[r * A_LDK + q * 8],
                 Kg + base + (size_t)(kt + r) * INNER + q * 8);
      union { uint4 u; _Float16 hh[8]; } tmp;
      tmp.u = *(const uint4*)(Vg + base + (size_t)(kt + r) * INNER + q * 8);
      #pragma unroll
      for (int d = 0; d < 8; ++d) Vt[(q * 8 + d) * A_LDK + r] = tmp.hh[d];
      if (kt + 64 < SEQ) {
        __builtin_prefetch(Kg + base + (size_t)(kt + 64 + r) * INNER + q * 8, 0, 0);
        __builtin_prefetch(Vg + base + (size_t)(kt + 64 + r) * INNER + q * 8, 0, 0);
      }
    }
    async_wait0();
    __syncthreads();

    // S = (Q*scale) K^T : 4 n-tiles x 2 k-chunks
    v8f s[4];
    #pragma unroll
    for (int j = 0; j < 4; ++j) {
      #pragma unroll
      for (int v = 0; v < 8; ++v) s[j][v] = 0.f;
      #pragma unroll
      for (int c = 0; c < 2; ++c) {
        v16h bf = lds_b_frag(Ks, A_LDK, j * 16, c * 32);
        s[j] = __builtin_amdgcn_wmma_f32_16x16x32_f16(
            false, qf[c], false, bf, (short)0, s[j], false, false);
      }
    }

    // row max of tile (reduce over frags, then over 16 lanes of each half)
    float tmax[8];
    #pragma unroll
    for (int v = 0; v < 8; ++v)
      tmax[v] = fmaxf(fmaxf(s[0][v], s[1][v]), fmaxf(s[2][v], s[3][v]));
    #pragma unroll
    for (int off = 1; off < 16; off <<= 1)
      #pragma unroll
      for (int v = 0; v < 8; ++v)
        tmax[v] = fmaxf(tmax[v], __shfl_xor(tmax[v], off, 32));

    float alpha[8];
    #pragma unroll
    for (int v = 0; v < 8; ++v) {
      const float mnew = fmaxf(mrow[v], tmax[v]);
      alpha[v] = __expf(mrow[v] - mnew);
      mrow[v] = mnew;
    }

    float tsum[8];
    #pragma unroll
    for (int v = 0; v < 8; ++v) tsum[v] = 0.f;
    #pragma unroll
    for (int j = 0; j < 4; ++j)
      #pragma unroll
      for (int v = 0; v < 8; ++v) {
        const float p = __expf(s[j][v] - mrow[v]);
        s[j][v] = p;
        tsum[v] += p;
      }
    #pragma unroll
    for (int off = 1; off < 16; off <<= 1)
      #pragma unroll
      for (int v = 0; v < 8; ++v)
        tsum[v] += __shfl_xor(tsum[v], off, 32);
    #pragma unroll
    for (int v = 0; v < 8; ++v) lrow[v] = lrow[v] * alpha[v] + tsum[v];

    #pragma unroll
    for (int j = 0; j < 4; ++j)
      #pragma unroll
      for (int v = 0; v < 8; ++v) o[j][v] *= alpha[v];

    // P (D-frag layout) -> per-wave LDS -> A-frag layout; DS is in-order per wave
    {
      const int rlo = (lane >> 4) * 8;
      const int col = lane & 15;
      #pragma unroll
      for (int j = 0; j < 4; ++j)
        #pragma unroll
        for (int v = 0; v < 8; ++v)
          myP[(rlo + v) * A_LDK + j * 16 + col] = (_Float16)s[j][v];
    }

    // O += P * V : 2 k-chunks (key) x 4 n-tiles (d)
    #pragma unroll
    for (int c = 0; c < 2; ++c) {
      v16h pf = lds_a_frag(myP, A_LDK, 0, c * 32);
      #pragma unroll
      for (int j = 0; j < 4; ++j) {
        v16h vf = lds_b_frag(Vt, A_LDK, j * 16, c * 32);
        o[j] = __builtin_amdgcn_wmma_f32_16x16x32_f16(
            false, pf, false, vf, (short)0, o[j], false, false);
      }
    }
  }

  // normalize and store
  {
    const int rlo = (lane >> 4) * 8;
    const int col = lane & 15;
    #pragma unroll
    for (int v = 0; v < 8; ++v) {
      const float inv = 1.f / lrow[v];
      _Float16* op = Og + base + (size_t)(qrow0 + rlo + v) * INNER;
      #pragma unroll
      for (int j = 0; j < 4; ++j)
        op[j * 16 + col] = (_Float16)(o[j][v] * inv);
    }
  }
}

// ---------------------------------------------------------------------------
extern "C" void kernel_launch(void* const* d_in, const int* in_sizes, int n_in,
                              void* d_out, int out_size, void* d_ws, size_t ws_size,
                              hipStream_t stream) {
  const float* x  = (const float*)d_in[0];
  const float* Wq = (const float*)d_in[1];
  const float* Wk = (const float*)d_in[2];
  const float* Wv = (const float*)d_in[3];
  const float* Wo = (const float*)d_in[4];
  const float* bo = (const float*)d_in[5];
  float* out = (float*)d_out;

  char* ws = (char*)d_ws;
  _Float16* xh  = (_Float16*)(ws);                          // 8 MB
  _Float16* WqT = (_Float16*)(ws + ((size_t)8  << 20));     // 2 MB each
  _Float16* WkT = (_Float16*)(ws + ((size_t)10 << 20));
  _Float16* WvT = (_Float16*)(ws + ((size_t)12 << 20));
  _Float16* WoT = (_Float16*)(ws + ((size_t)14 << 20));
  _Float16* Qh  = (_Float16*)(ws + ((size_t)16 << 20));     // 8 MB each
  _Float16* Kh  = (_Float16*)(ws + ((size_t)24 << 20));
  _Float16* Vh  = (_Float16*)(ws + ((size_t)32 << 20));
  _Float16* Oh  = (_Float16*)(ws + ((size_t)40 << 20));

  const int M = 2 * SEQ, N = INNER, Kd = 1024;

  cvt_f16<<<(M * Kd) / 256, 256, 0, stream>>>(x, xh, M * Kd);
  dim3 tg(Kd / 32, N / 32);
  transpose_cvt<<<tg, 256, 0, stream>>>(Wq, WqT, Kd, N);
  transpose_cvt<<<tg, 256, 0, stream>>>(Wk, WkT, Kd, N);
  transpose_cvt<<<tg, 256, 0, stream>>>(Wv, WvT, Kd, N);
  transpose_cvt<<<tg, 256, 0, stream>>>(Wo, WoT, Kd, N);

  dim3 gg(M / 128, N / 128);
  gemm_tn<false><<<gg, 256, 0, stream>>>(xh, WqT, Qh, nullptr, M, N, Kd);
  gemm_tn<false><<<gg, 256, 0, stream>>>(xh, WkT, Kh, nullptr, M, N, Kd);
  gemm_tn<false><<<gg, 256, 0, stream>>>(xh, WvT, Vh, nullptr, M, N, Kd);

  flash_attn<<<2 * 16 * (SEQ / 128), 256, 0, stream>>>(Qh, Kh, Vh, Oh);

  gemm_tn<true><<<gg, 256, 0, stream>>>(Oh, WoT, out, bo, M, N, Kd);
}
